// AttentionModel_23991687315554
// MI455X (gfx1250) — matrix operand order repaired and checked by reading, not compiled, not run on hardware
//
#include <hip/hip_runtime.h>

typedef __attribute__((ext_vector_type(16))) _Float16 v16h;
typedef __attribute__((ext_vector_type(8)))  float    v8f;

#define NPIX 4096
#define CCH  256
#define C8   32

union Frag {
    v16h v;
    uint4 q[2];
    _Float16 h[16];
};

union H4 {
    uint2 u;
    _Float16 h[4];
};

static __device__ __forceinline__ v8f wmma16(const Frag& a, const Frag& b, v8f c) {
    return __builtin_amdgcn_wmma_f32_16x16x32_f16(false, a.v, false, b.v, (short)0, c,
                                                  false, false);
}

static __device__ __forceinline__ const float* wrow_for(int o, const float* Wq,
                                                        const float* Wk, const float* Wv) {
    if (o < 32) return Wq + o * CCH;
    if (o < 64) return Wk + (o - 32) * CCH;
    return Wv + (o - 64) * CCH;
}
static __device__ __forceinline__ float bias_for(int o, const float* bq,
                                                 const float* bk, const float* bv) {
    if (o < 32) return bq[o];
    if (o < 64) return bk[o - 32];
    return bv[o - 64];
}
static __device__ __forceinline__ _Float16* drow_for(int o, int b, _Float16* qc,
                                                     _Float16* kc, _Float16* vc) {
    if (o < 32) return qc + ((size_t)b * C8 + o) * NPIX;
    if (o < 64) return kc + ((size_t)b * C8 + (o - 32)) * NPIX;
    return vc + ((size_t)b * CCH + (o - 64)) * NPIX;
}

// ---------------------------------------------------------------------------
// Kernel 1: fused q/k/v 1x1-conv projections as one [320 x 256] x [256 x N]
// GEMM per batch, f16 WMMA with fp32 accumulate. Outputs channel-major f16.
// Grid: 8 batches x 64 n-chunks (64 cols each). Block: 640 threads (20 waves);
// wave w owns ONE 16-row M-tile (o = 16w..16w+15), so A-fragments are only
// 64 VGPRs/wave and nothing spills.
// ---------------------------------------------------------------------------
__global__ __launch_bounds__(640) void proj_kernel(
    const float* __restrict__ x,
    const float* __restrict__ Wq, const float* __restrict__ bq,
    const float* __restrict__ Wk, const float* __restrict__ bk,
    const float* __restrict__ Wv, const float* __restrict__ bv,
    _Float16* __restrict__ qc, _Float16* __restrict__ kc, _Float16* __restrict__ vc)
{
    __shared__ _Float16 xs[64 * 264];   // [n][c] transposed, row stride 264 halves

    const int tid = threadIdx.x;
    const int b  = blockIdx.x >> 6;
    const int n0 = (blockIdx.x & 63) * 64;
    const size_t xbase = (size_t)b * CCH * NPIX;

    // Stage x tile [c=256][n=64] -> xs[n][c] in f16 (transpose).
    for (int e = tid; e < 4096; e += 640) {
        const int c  = e >> 4;
        const int f4 = e & 15;
        const float4 v4 = *(const float4*)(x + xbase + (size_t)c * NPIX + n0 + f4 * 4);
        const int nn = f4 * 4;
        xs[(nn + 0) * 264 + c] = (_Float16)v4.x;
        xs[(nn + 1) * 264 + c] = (_Float16)v4.y;
        xs[(nn + 2) * 264 + c] = (_Float16)v4.z;
        xs[(nn + 3) * 264 + c] = (_Float16)v4.w;
    }
    __syncthreads();

    const int w     = tid >> 5;
    const int lane  = tid & 31;
    const int l16   = lane & 15;
    const int lhalf = lane >> 4;

    // A fragments: W rows in f16, A layout 16x32 (lane<16: K {0-7,16-23}).
    Frag af[8];
    {
        const int o = w * 16 + l16;
        const float* wr = wrow_for(o, Wq, Wk, Wv);
        #pragma unroll
        for (int kc8 = 0; kc8 < 8; ++kc8) {
            const int cb = kc8 * 32 + lhalf * 8;
            const float4 f0 = *(const float4*)(wr + cb);
            const float4 f1 = *(const float4*)(wr + cb + 4);
            const float4 f2 = *(const float4*)(wr + cb + 16);
            const float4 f3 = *(const float4*)(wr + cb + 20);
            Frag& fr = af[kc8];
            fr.h[0]=(_Float16)f0.x; fr.h[1]=(_Float16)f0.y; fr.h[2]=(_Float16)f0.z; fr.h[3]=(_Float16)f0.w;
            fr.h[4]=(_Float16)f1.x; fr.h[5]=(_Float16)f1.y; fr.h[6]=(_Float16)f1.z; fr.h[7]=(_Float16)f1.w;
            fr.h[8]=(_Float16)f2.x; fr.h[9]=(_Float16)f2.y; fr.h[10]=(_Float16)f2.z; fr.h[11]=(_Float16)f2.w;
            fr.h[12]=(_Float16)f3.x; fr.h[13]=(_Float16)f3.y; fr.h[14]=(_Float16)f3.z; fr.h[15]=(_Float16)f3.w;
        }
    }

    // Per-output-row bias and destination pointers (rows o = 16w + lhalf*8 + r).
    float     biasr[8];
    _Float16* drow[8];
    #pragma unroll
    for (int r = 0; r < 8; ++r) {
        const int o = w * 16 + lhalf * 8 + r;
        biasr[r] = bias_for(o, bq, bk, bv);
        drow[r]  = drow_for(o, b, qc, kc, vc);
    }

    #pragma unroll 1
    for (int nt = 0; nt < 4; ++nt) {
        v8f acc = {};
        #pragma unroll
        for (int kc8 = 0; kc8 < 8; ++kc8) {
            Frag bf;  // B layout 32x16: lane<16 holds K {0-7,16-23} of its column
            const _Float16* base = xs + (size_t)(nt * 16 + l16) * 264 + kc8 * 32 + lhalf * 8;
            bf.q[0] = *(const uint4*)(base);
            bf.q[1] = *(const uint4*)(base + 16);
            acc = wmma16(af[kc8], bf, acc);
        }
        const int n = n0 + nt * 16 + l16;
        #pragma unroll
        for (int r = 0; r < 8; ++r) {
            drow[r][n] = (_Float16)(acc[r] + biasr[r]);
        }
    }
}

// ---------------------------------------------------------------------------
// Kernel 2: flash attention with swapped roles (Q<-k, K<-q, V<-v),
// softmax over i without max-subtraction (safe for |S| <~ 40 in fp32).
// Grid: 8 batches x 32 j-blocks of 128. Block: 256 threads = 8 waves,
// wave w handles j-tile [j0 + 16w, +16), accumulates O[16 x 256] in VGPRs.
// ---------------------------------------------------------------------------
__global__ __launch_bounds__(256) void attn_kernel(
    const _Float16* __restrict__ qc, const _Float16* __restrict__ kc,
    const _Float16* __restrict__ vc,
    const float* __restrict__ x, const float* __restrict__ gamma_p,
    float* __restrict__ out)
{
    __shared__ _Float16 sh[21760];
    _Float16* kbuf = sh;            // [128][40]  k^T staged, row stride 40
    _Float16* vbuf = sh + 5120;     // [256][40]  v chunk, [c][i-local]
    _Float16* qbuf = sh + 15360;    // [32][40]   q chunk transposed, [i-local][o]
    _Float16* pbuf = sh + 16640;    // 8 waves x [16][40] P tiles

    const int tid   = threadIdx.x;
    const int b     = blockIdx.x >> 5;
    const int j0    = (blockIdx.x & 31) * 128;
    const int w     = tid >> 5;
    const int lane  = tid & 31;
    const int l16   = lane & 15;
    const int lhalf = lane >> 4;

    const size_t qkbase = (size_t)b * C8 * NPIX;
    const size_t vbase  = (size_t)b * CCH * NPIX;

    // Stage kbuf[j][o] = kc[b][o][j0+j] (transpose).
    for (int e = tid; e < 1024; e += 256) {
        const int o  = e >> 5;
        const int jq = (e & 31) * 4;
        H4 d; d.u = *(const uint2*)(kc + qkbase + (size_t)o * NPIX + j0 + jq);
        kbuf[(jq + 0) * 40 + o] = d.h[0];
        kbuf[(jq + 1) * 40 + o] = d.h[1];
        kbuf[(jq + 2) * 40 + o] = d.h[2];
        kbuf[(jq + 3) * 40 + o] = d.h[3];
    }
    __syncthreads();

    // A_k: k^T tile for this wave's 16 rows j, K = all 32 channels.
    Frag ak;
    {
        const _Float16* base = kbuf + (size_t)(w * 16 + l16) * 40 + lhalf * 8;
        ak.q[0] = *(const uint4*)(base);
        ak.q[1] = *(const uint4*)(base + 16);
    }

    v8f acc[16];
    #pragma unroll
    for (int t = 0; t < 16; ++t) { v8f z = {}; acc[t] = z; }
    float sl[8];
    #pragma unroll
    for (int r = 0; r < 8; ++r) sl[r] = 0.0f;

    _Float16* pw = pbuf + w * 640;   // this wave's P tile [16][40]

    #pragma unroll 1
    for (int i0 = 0; i0 < NPIX; i0 += 32) {
        if (i0 + 32 < NPIX)
            __builtin_prefetch(vc + vbase + (size_t)tid * NPIX + i0 + 32, 0, 3);
        __syncthreads();
        // Stage v chunk: vbuf[c][k] = vc[b][c][i0+k]  (layout already matches)
        for (int e = tid; e < 1024; e += 256) {
            const int c  = e >> 2;
            const int kq = (e & 3) * 8;
            const uint4 d = *(const uint4*)(vc + vbase + (size_t)c * NPIX + i0 + kq);
            *(uint4*)(vbuf + (size_t)c * 40 + kq) = d;
        }
        // Stage q chunk transposed: qbuf[i][o] = qc[b][o][i0+i]
        {
            const int o  = tid >> 3;
            const int iq = (tid & 7) * 4;
            H4 d; d.u = *(const uint2*)(qc + qkbase + (size_t)o * NPIX + i0 + iq);
            qbuf[(iq + 0) * 40 + o] = d.h[0];
            qbuf[(iq + 1) * 40 + o] = d.h[1];
            qbuf[(iq + 2) * 40 + o] = d.h[2];
            qbuf[(iq + 3) * 40 + o] = d.h[3];
        }
        __syncthreads();

        // B_q fragments for i-chunk halves.
        Frag bq0, bq1;
        {
            const _Float16* b0 = qbuf + (size_t)l16 * 40 + lhalf * 8;
            bq0.q[0] = *(const uint4*)(b0);
            bq0.q[1] = *(const uint4*)(b0 + 16);
            const _Float16* b1 = qbuf + (size_t)(16 + l16) * 40 + lhalf * 8;
            bq1.q[0] = *(const uint4*)(b1);
            bq1.q[1] = *(const uint4*)(b1 + 16);
        }
        v8f z = {};
        v8f s0 = wmma16(ak, bq0, z);   // S[j, i0..i0+15]
        v8f s1 = wmma16(ak, bq1, z);   // S[j, i0+16..i0+31]

        // P = exp(S); partial row sums; write P tile (C/D -> memory [j][i]).
        const int jrow = lhalf * 8;
        #pragma unroll
        for (int r = 0; r < 8; ++r) {
            const float p0 = __expf(s0[r]);
            const float p1 = __expf(s1[r]);
            sl[r] += p0 + p1;
            pw[(jrow + r) * 40 + l16]      = (_Float16)p0;
            pw[(jrow + r) * 40 + 16 + l16] = (_Float16)p1;
        }
        // Reload P as A fragment (same-wave LDS ops are in-order).
        Frag ap;
        {
            const _Float16* base = pw + (size_t)l16 * 40 + lhalf * 8;
            ap.q[0] = *(const uint4*)(base);
            ap.q[1] = *(const uint4*)(base + 16);
        }
        // O[j, c] += P[j, i] * v[c, i] over the 16 c-tiles.
        #pragma unroll
        for (int t = 0; t < 16; ++t) {
            Frag bv_;
            const _Float16* base = vbuf + (size_t)(t * 16 + l16) * 40 + lhalf * 8;
            bv_.q[0] = *(const uint4*)(base);
            bv_.q[1] = *(const uint4*)(base + 16);
            acc[t] = wmma16(ap, bv_, acc[t]);
        }
    }

    // Finish row sums: butterfly within each 16-lane half, then invert.
    #pragma unroll
    for (int r = 0; r < 8; ++r) {
        float s = sl[r];
        s += __shfl_xor(s, 1);
        s += __shfl_xor(s, 2);
        s += __shfl_xor(s, 4);
        s += __shfl_xor(s, 8);
        sl[r] = 1.0f / s;
    }

    // Epilogue: out[b][c][j] = gamma * O[j][c] / l[j] + x[b][c][j],
    // transposed through this wave's LDS region for coalesced 32B stores.
    const float gamma = *gamma_p;
    float* T = (float*)pw;          // [16][17] fp32, fits in 1280B wave region
    const int j0w = j0 + w * 16;
    const int cl  = lane >> 1;
    const int jg  = lane & 1;
    #pragma unroll 1
    for (int t = 0; t < 16; ++t) {
        #pragma unroll
        for (int r = 0; r < 8; ++r)
            T[(lhalf * 8 + r) * 17 + l16] = acc[t][r] * sl[r];
        // same-wave LDS in-order: safe to read back
        const int c = t * 16 + cl;
        const size_t obase = ((size_t)b * CCH + c) * NPIX + j0w + jg * 8;
        const float4 x0 = *(const float4*)(x + obase);
        const float4 x1 = *(const float4*)(x + obase + 4);
        float4 r0, r1;
        r0.x = gamma * T[(jg * 8 + 0) * 17 + cl] + x0.x;
        r0.y = gamma * T[(jg * 8 + 1) * 17 + cl] + x0.y;
        r0.z = gamma * T[(jg * 8 + 2) * 17 + cl] + x0.z;
        r0.w = gamma * T[(jg * 8 + 3) * 17 + cl] + x0.w;
        r1.x = gamma * T[(jg * 8 + 4) * 17 + cl] + x1.x;
        r1.y = gamma * T[(jg * 8 + 5) * 17 + cl] + x1.y;
        r1.z = gamma * T[(jg * 8 + 6) * 17 + cl] + x1.z;
        r1.w = gamma * T[(jg * 8 + 7) * 17 + cl] + x1.w;
        *(float4*)(out + obase)     = r0;
        *(float4*)(out + obase + 4) = r1;
    }
}

extern "C" void kernel_launch(void* const* d_in, const int* in_sizes, int n_in,
                              void* d_out, int out_size, void* d_ws, size_t ws_size,
                              hipStream_t stream) {
    (void)in_sizes; (void)n_in; (void)out_size; (void)ws_size;
    const float* x     = (const float*)d_in[0];
    const float* Wq    = (const float*)d_in[1];
    const float* bq    = (const float*)d_in[2];
    const float* Wk    = (const float*)d_in[3];
    const float* bk    = (const float*)d_in[4];
    const float* Wv    = (const float*)d_in[5];
    const float* bv    = (const float*)d_in[6];
    const float* gamma = (const float*)d_in[7];
    float* out = (float*)d_out;

    char* ws = (char*)d_ws;                               // 20 MB used
    _Float16* qc = (_Float16*)(ws);                       // [8][32][4096]  2 MB
    _Float16* kc = (_Float16*)(ws + (size_t)2 * 1024 * 1024);  // 2 MB
    _Float16* vc = (_Float16*)(ws + (size_t)4 * 1024 * 1024);  // [8][256][4096] 16 MB

    proj_kernel<<<512, 640, 0, stream>>>(x, Wq, bq, Wk, bk, Wv, bv, qc, kc, vc);
    attn_kernel<<<256, 256, 0, stream>>>(qc, kc, vc, x, gamma, out);
}
